// MTMSGRU_57088705298492
// MI455X (gfx1250) — matrix operand order repaired and checked by reading, not compiled, not run on hardware
//
#include <hip/hip_runtime.h>

// ---------------- problem constants ----------------
#define BB   64
#define NN   1085
#define HIST 24
#define PRED 24
#define KHOP 3
#define HH   64          // GRU hidden
#define FXX  1
#define FMM  11
#define FF   12          // FX + FM
#define EE   17360
#define BN   (BB * NN)   // 69440  (divisible by 16)
#define G3   (3 * HH)    // 192
#define K2F  (2 * FF)    // 24

typedef __attribute__((ext_vector_type(16))) __bf16 v16bf;
typedef __attribute__((ext_vector_type(8)))  float  v8f;

__device__ __forceinline__ unsigned short f2bf(float f) {
  union { float f; unsigned int u; } v; v.f = f;
  unsigned int u = v.u;
  unsigned int lsb = (u >> 16) & 1u;
  u += 0x7FFFu + lsb;             // round-to-nearest-even
  return (unsigned short)(u >> 16);
}

// fast transcendentals: v_exp_f32 + v_rcp_f32 instead of IEEE division chains
__device__ __forceinline__ float fast_sigmoid(float x) {
  return __builtin_amdgcn_rcpf(1.0f + __expf(-x));      // x->-inf: rcp(inf)=0 OK
}
__device__ __forceinline__ float fast_tanh(float x) {
  // tanh(x) = 1 - 2/(e^{2x}+1); saturates correctly at +/-inf without inf*0
  return 1.0f - 2.0f * __builtin_amdgcn_rcpf(__expf(2.0f * x) + 1.0f);
}

// ---------------- weight packing into WMMA B-operand lane order ----------------
// B packing here (bf16, 16x16x32, contiguous k matching the A gather): lane<16 -> K=elt (0..15), lane>=16 -> K=16+elt; N = lane&15.
// WxP layout: [k][ntile(12)][lane(32)][elt(16)]  (K rows >= 24 zero-padded)
// WhP layout: [k][ntile*2+kslab(24)][lane(32)][elt(16)]
__global__ __launch_bounds__(256) void pack_w_kernel(
    const float* __restrict__ Wx, const float* __restrict__ Wh,
    unsigned short* __restrict__ WxP, unsigned short* __restrict__ WhP) {
  int tid = blockIdx.x * 256 + threadIdx.x;
  const int nWx = KHOP * 12 * 512;
  if (tid < nWx) {
    int k    = tid / (12 * 512);
    int rem  = tid % (12 * 512);
    int nt   = rem / 512;
    int lane = (rem & 511) >> 4;
    int i    = rem & 15;
    int row  = (lane < 16) ? i : (16 + i);
    int col  = nt * 16 + (lane & 15);
    float v  = (row < K2F) ? Wx[(k * K2F + row) * G3 + col] : 0.0f;
    WxP[tid] = f2bf(v);
  }
  const int nWh = KHOP * 24 * 512;
  if (tid < nWh) {
    int k    = tid / (24 * 512);
    int rem  = tid % (24 * 512);
    int tile = rem / 512;
    int nt   = tile >> 1, ks = tile & 1;
    int lane = (rem & 511) >> 4;
    int i    = rem & 15;
    int row  = ks * 32 + ((lane < 16) ? i : (16 + i));
    int col  = nt * 16 + (lane & 15);
    WhP[tid] = f2bf(Wh[(k * HH + row) * G3 + col]);
  }
}

// ---------------- misc small kernels ----------------
__global__ __launch_bounds__(256) void zero_kernel(float4* __restrict__ p, int n4) {
  int tid = blockIdx.x * 256 + threadIdx.x;
  if (tid < n4) p[tid] = make_float4(0.f, 0.f, 0.f, 0.f);
}

__global__ __launch_bounds__(256) void copy_kernel(float4* __restrict__ d,
                                                   const float4* __restrict__ s, int n4) {
  int tid = blockIdx.x * 256 + threadIdx.x;
  if (tid < n4) d[tid] = s[tid];
}

// xn[b,n,0] = xbase[b*xstr + n]; xn[b,n,1..11] = mbase[b*mstr + n*FM + j]
__global__ __launch_bounds__(256) void assemble_kernel(
    float* __restrict__ xn, const float* __restrict__ xbase, int xstr,
    const float* __restrict__ mbase, int mstr) {
  int tid = blockIdx.x * 256 + threadIdx.x;
  if (tid >= BN) return;
  int b = tid / NN, n = tid - b * NN;
  float* o = xn + tid * FF;
  o[0] = xbase[b * xstr + n];
  const float* m = mbase + b * mstr + n * FMM;
#pragma unroll
  for (int j = 0; j < FMM; j++) o[1 + j] = m[j];
}

// edge MLP + scatter:  acc[b,dst[e],:] += relu([x[b,src]||x[b,dst]] @ We + be)
__global__ __launch_bounds__(256) void edge_kernel(
    const float* __restrict__ x, float* __restrict__ acc,
    const int* __restrict__ src, const int* __restrict__ dst,
    const float* __restrict__ We, const float* __restrict__ be) {
  __shared__ float sW[K2F * FF + FF];       // 288 + 12
  for (int i = threadIdx.x; i < K2F * FF + FF; i += 256)
    sW[i] = (i < K2F * FF) ? We[i] : be[i - K2F * FF];
  __syncthreads();
  int tid = blockIdx.x * 256 + threadIdx.x; // B*E = 4340*256 exactly
  int b = tid / EE, e = tid - b * EE;
  int sI = src[e], dI = dst[e];
  const float* xs = x + (b * NN + sI) * FF;
  const float* xd = x + (b * NN + dI) * FF;
  float xc[K2F];
#pragma unroll
  for (int j = 0; j < FF; j++) { xc[j] = xs[j]; xc[FF + j] = xd[j]; }
  float* ap = acc + (b * NN + dI) * FF;
#pragma unroll
  for (int f = 0; f < FF; f++) {
    float m = sW[K2F * FF + f];
#pragma unroll
    for (int j = 0; j < K2F; j++) m += xc[j] * sW[j * FF + f];  // LDS broadcast reads
    m = fmaxf(m, 0.0f);
    atomicAdd(ap + f, m);                   // global_atomic_add_f32, L2-resident
  }
}

// ---------------- fused GRU step (WMMA bf16) ----------------
// Block = 128 threads (4 wave32) handles 16 rows.  g = [xg||xn] @ Wx + bx (K=24 pad 32),
// gh = h @ Wh + bh (K=64 = 2x32), then gates + in-place h update.
__global__ __launch_bounds__(128) void gru_kernel(
    const float* __restrict__ xg, const float* __restrict__ xnf,
    float* __restrict__ h,
    const unsigned short* __restrict__ WxP, const unsigned short* __restrict__ WhP,
    const float* __restrict__ bx, const float* __restrict__ bh) {
  __shared__ float lxa[16][32];   // x_gru rows, K padded to 32
  __shared__ float lha[16][64];   // h rows
  __shared__ float lg[16][192];   // g
  __shared__ float lgh[16][192];  // gh
  const int t  = threadIdx.x;
  const int r0 = blockIdx.x * 16;

  for (int i = t; i < 16 * 32; i += 128) {
    int row = i >> 5, c = i & 31;
    float v = 0.0f;
    if (c < FF)          v = xg [(r0 + row) * FF + c];
    else if (c < 2 * FF) v = xnf[(r0 + row) * FF + (c - FF)];
    lxa[row][c] = v;
  }
  for (int i = t; i < 16 * 64; i += 128) {
    int row = i >> 6, c = i & 63;
    lha[row][c] = h[(r0 + row) * HH + c];
  }
  __syncthreads();

  const int wave = t >> 5;        // wave-uniform
  const int lane = t & 31;
  const int mrow = lane & 15;     // A row / D column index
  const int hi   = lane >> 4;

  for (int nt = wave; nt < 12; nt += 4) {   // uniform per wave -> EXEC all ones
    const int ncol = nt * 16 + mrow;
    // ---- g tile: A = x_gru (16x32 bf16 gather, contiguous k = 16*hi + i) ----
    v16bf a;
#pragma unroll
    for (int i = 0; i < 16; i++) {
      int kk = i + hi * 16;
      a[i] = (__bf16)lxa[mrow][kk];
    }
    v16bf bmat = *(const v16bf*)(WxP + (nt * 32 + lane) * 16);
    v8f c = {0.f, 0.f, 0.f, 0.f, 0.f, 0.f, 0.f, 0.f};
    c = __builtin_amdgcn_wmma_f32_16x16x32_bf16(false, a, false, bmat, (short)0, c,
                                                false, false);
    // ---- gh tile: K = 64 as two 32-slabs ----
    v8f cz = {0.f, 0.f, 0.f, 0.f, 0.f, 0.f, 0.f, 0.f};
#pragma unroll
    for (int ks = 0; ks < 2; ks++) {
      v16bf ah;
#pragma unroll
      for (int i = 0; i < 16; i++) {
        int kk = i + hi * 16 + ks * 32;
        ah[i] = (__bf16)lha[mrow][kk];
      }
      v16bf bh2 = *(const v16bf*)(WhP + ((nt * 2 + ks) * 32 + lane) * 16);
      cz = __builtin_amdgcn_wmma_f32_16x16x32_bf16(false, ah, false, bh2, (short)0,
                                                   cz, false, false);
    }
    float bxv = bx[ncol];
    float bhv = bh[ncol];
#pragma unroll
    for (int j = 0; j < 8; j++) {           // D: M = j + 8*hi, N = lane&15
      lg [j + 8 * hi][ncol] = c [j] + bxv;
      lgh[j + 8 * hi][ncol] = cz[j] + bhv;
    }
  }
  __syncthreads();

  for (int i = t; i < 16 * 64; i += 128) {  // gates
    int m = i >> 6, cc = i & 63;
    float r = fast_sigmoid(lg[m][cc]      + lgh[m][cc]);
    float z = fast_sigmoid(lg[m][64 + cc] + lgh[m][64 + cc]);
    float n = fast_tanh(lg[m][128 + cc] + r * lgh[m][128 + cc]);
    float hv = lha[m][cc];
    h[(r0 + m) * HH + cc] = (1.0f - z) * n + z * hv;
  }
}

// x_new = concat(h0,h1,h2) @ fc_w + fc_b ;  write xpred and output slice
__global__ __launch_bounds__(256) void fc_kernel(
    const float* __restrict__ h0, const float* __restrict__ h1,
    const float* __restrict__ h2, const float* __restrict__ fcw,
    const float* __restrict__ fcb, float* __restrict__ xpred,
    float* __restrict__ out, int ostr) {
  int tid = blockIdx.x * 256 + threadIdx.x;
  if (tid >= BN) return;
  int b = tid / NN, n = tid - b * NN;
  const float* p0 = h0 + tid * HH;
  const float* p1 = h1 + tid * HH;
  const float* p2 = h2 + tid * HH;
  float acc = fcb[0];
#pragma unroll 8
  for (int j = 0; j < HH; j++)
    acc += p0[j] * fcw[j] + p1[j] * fcw[HH + j] + p2[j] * fcw[2 * HH + j];
  xpred[tid]        = acc;
  out[b * ostr + n] = acc;
}

// ---------------- host orchestration (graph-captured) ----------------
extern "C" void kernel_launch(void* const* d_in, const int* in_sizes, int n_in,
                              void* d_out, int out_size, void* d_ws, size_t ws_size,
                              hipStream_t stream) {
  (void)in_sizes; (void)n_in; (void)out_size; (void)ws_size;
  const float* x_hist   = (const float*)d_in[0];
  const float* enc_misc = (const float*)d_in[1];
  const float* dec      = (const float*)d_in[2];
  const float* We       = (const float*)d_in[3];
  const float* be       = (const float*)d_in[4];
  const float* Wx       = (const float*)d_in[5];
  const float* Wh       = (const float*)d_in[6];
  const float* bx       = (const float*)d_in[7];
  const float* bh       = (const float*)d_in[8];
  const float* fc_w     = (const float*)d_in[9];
  const float* fc_b     = (const float*)d_in[10];
  const int*   srcI     = (const int*)d_in[11];
  const int*   dstI     = (const int*)d_in[12];
  float*       out      = (float*)d_out;

  size_t off = 0;
  char* base = (char*)d_ws;
  auto carve = [&](size_t bytes) -> char* {
    char* p = base + off;
    off += (bytes + 255) & ~(size_t)255;
    return p;
  };
  float* xn    = (float*)carve(sizeof(float) * BN * FF);
  float* xA    = (float*)carve(sizeof(float) * BN * FF);
  float* xB    = (float*)carve(sizeof(float) * BN * FF);
  float* hbuf  = (float*)carve(sizeof(float) * 3 * (size_t)BN * HH);
  float* xpred = (float*)carve(sizeof(float) * BN);
  unsigned short* WxP = (unsigned short*)carve(sizeof(unsigned short) * KHOP * 12 * 512);
  unsigned short* WhP = (unsigned short*)carve(sizeof(unsigned short) * KHOP * 24 * 512);

  const int abl = (BN + 255) / 256;                 // 272
  const int cbl = (BN * FF / 4 + 255) / 256;        // 814
  const int ebl = (BB * EE) / 256;                  // 4340 (exact)
  const int gbl = BN / 16;                          // 4340 (exact)
  const int h4  = 3 * BN * HH / 4;
  const int zbl = (h4 + 255) / 256;

  pack_w_kernel<<<(KHOP * 24 * 512 + 255) / 256, 256, 0, stream>>>(Wx, Wh, WxP, WhP);
  zero_kernel<<<zbl, 256, 0, stream>>>((float4*)hbuf, h4);

  float* hk[3] = {hbuf, hbuf + (size_t)BN * HH, hbuf + 2 * (size_t)BN * HH};

  for (int step = 0; step < (HIST - 1) + PRED; step++) {
    const bool pred = step >= (HIST - 1);
    const int  s    = step - (HIST - 1);
    const float* xbase; int xstr;
    const float* mbase; int mstr;
    if (!pred) {
      xbase = x_hist + (size_t)step * NN;               xstr = HIST * NN;
      mbase = enc_misc + (size_t)(step + 1) * NN * FMM; mstr = HIST * NN * FMM;
    } else {
      if (s == 0) { xbase = x_hist + (size_t)(HIST - 1) * NN; xstr = HIST * NN; }
      else        { xbase = xpred;                            xstr = NN; }
      mbase = dec + (size_t)s * NN * FMM;               mstr = PRED * NN * FMM;
    }
    assemble_kernel<<<abl, 256, 0, stream>>>(xn, xbase, xstr, mbase, mstr);

    const float* cur = xn;
    float* bufs[2] = {xA, xB};
    for (int k = 0; k < KHOP; k++) {
      float* nb = bufs[k & 1];
      copy_kernel<<<cbl, 256, 0, stream>>>((float4*)nb, (const float4*)cur, BN * FF / 4);
      edge_kernel<<<ebl, 256, 0, stream>>>(cur, nb, srcI, dstI, We, be);
      cur = nb;
      gru_kernel<<<gbl, 128, 0, stream>>>(cur, xn, hk[k],
                                          WxP + (size_t)k * 12 * 512,
                                          WhP + (size_t)k * 24 * 512,
                                          bx + k * G3, bh + k * G3);
    }
    if (pred) {
      fc_kernel<<<abl, 256, 0, stream>>>(hk[0], hk[1], hk[2], fc_w, fc_b,
                                         xpred, out + (size_t)s * NN, PRED * NN);
    }
  }
}
